// HANLayer_1185410973815
// MI455X (gfx1250) — matrix operand order repaired and checked by reading, not compiled, not run on hardware
//
#include <hip/hip_runtime.h>
#include <hip/hip_bf16.h>

// ---------------- problem constants ----------------
#define NN   50000     // nodes
#define EE   800000    // edges per metapath
#define MM   3         // metapaths
#define FF   256       // H*D = in_dim = out feature
#define HH   8         // heads
#define DD   32        // per-head dim
#define HIDN 128       // semantic hidden
#define NEG_SLOPE 0.2f

typedef __attribute__((ext_vector_type(4)))  __bf16 v4bf;
typedef __attribute__((ext_vector_type(8)))  __bf16 v8bf;
typedef __attribute__((ext_vector_type(16))) __bf16 v16bf;
typedef __attribute__((ext_vector_type(8)))  float  v8f;

// monotone float<->uint encoding so atomicMax(uint) == float max
__device__ __forceinline__ unsigned fenc(float f) {
    unsigned b = __float_as_uint(f);
    return b ^ ((((int)b) >> 31) | 0x80000000u);
}
__device__ __forceinline__ float fdec(unsigned k) {
    unsigned b = (k & 0x80000000u) ? (k ^ 0x80000000u) : ~k;
    return __uint_as_float(b);
}
__device__ __forceinline__ float lrelu(float x) { return x > 0.f ? x : NEG_SLOPE * x; }

// ---------------------------------------------------------------------------
// K0b: convert h [NN,256] fp32 -> bf16 once (25.6MB, stays L2-resident).
// ---------------------------------------------------------------------------
__global__ void conv_h_bf16(const float* __restrict__ h, __bf16* __restrict__ hb) {
    long gid = (long)blockIdx.x * blockDim.x + threadIdx.x;   // over NN*FF/4
    if (gid >= (long)NN * FF / 4) return;
    float4 v = ((const float4*)h)[gid];
    v4bf o;
    o[0] = (__bf16)v.x; o[1] = (__bf16)v.y; o[2] = (__bf16)v.z; o[3] = (__bf16)v.w;
    *(v4bf*)(hb + 4 * gid) = o;
}

// ---------------------------------------------------------------------------
// K1: feat[m] = h @ fc_w[m]   via v_wmma_f32_16x16x32_bf16
//
// Block = 256 threads = 8 waves. Each block owns one (metapath m, 64-col
// panel); the 256x64 fp32 B-panel is cooperatively loaded, converted to bf16
// and stored to LDS PRE-SWIZZLED into WMMA B-fragment order, so each lane's
// 16 bf16 fragment is one contiguous 32B LDS read. Each wave computes a
// 16x64 output tile (4 accumulators) reusing one A fragment per k-step
// (4 WMMA per A load). A is read from the bf16 copy: two 16B loads/step.
//
// Fragment layouts (bf16; A/B use this code's contiguous k order, not ISA table order; C/D per ISA):
//  A 16x32: lanes 0-15 row=lane, K 0..15; lanes 16-31 K 16..31
//  B 32x16: lane col = lane&15; lanes 0-15 hold K kk+0..15, lanes 16-31
//           hold K kk+16..31 (b[0..15] contiguous in K)
//  C/D:     VGPR v -> row = tile*16 + v + (lane>=16 ? 8 : 0), col = lane&15
// ---------------------------------------------------------------------------
__global__ void gemm_feat_wmma(const __bf16* __restrict__ hb,
                               const float* __restrict__ fcw,
                               float* __restrict__ feat) {
    __shared__ __bf16 Bp[16384];            // 8 kchunks x 4 tiles x 32 lanes x 16 = 32KB
    const int tid  = threadIdx.x;
    const int lane = tid & 31, wave = tid >> 5;
    const int m       = blockIdx.z;         // 3
    const int panel   = blockIdx.y;         // 4 panels of 64 cols
    const int rowTile = blockIdx.x * 8 + wave;

    // ---- cooperative B-panel stage: fp32 -> bf16, swizzled to fragment order
    const float* B = fcw + (size_t)m * FF * FF + panel * 64;
#pragma unroll 4
    for (int it = 0; it < 64; ++it) {
        int e = it * 256 + tid;             // 0..16383
        int k = e >> 6, c = e & 63;         // row k of panel, col c in panel
        float x = B[(size_t)k * FF + c];
        int kc  = k >> 5;                   // k-chunk (32 wide)
        int rem = k & 31;
        int L   = ((rem >> 4) << 4) | (c & 15);   // owning lane
        int idx = rem & 15;                 // position in lane's 16-elem run
        int t   = c >> 4;                   // 16-col tile within panel
        Bp[(((kc << 2) + t) << 9) + (L << 4) + idx] = (__bf16)x;
    }
    __syncthreads();

    if (rowTile < 3125) {
        const int rowA  = rowTile * 16 + (lane & 15);
        const int kgrpA = (lane >> 4) * 16;
        v8f acc0 = {}, acc1 = {}, acc2 = {}, acc3 = {};
        for (int kc = 0; kc < 8; ++kc) {
            // A fragment: two contiguous 16B bf16 loads
            const __bf16* ap = hb + (size_t)rowA * FF + kc * 32 + kgrpA;
            v8bf a0 = *(const v8bf*)ap;
            v8bf a1 = *(const v8bf*)(ap + 8);
            v16bf a;
#pragma unroll
            for (int i = 0; i < 8; ++i) { a[i] = a0[i]; a[8 + i] = a1[i]; }
            // B fragments: contiguous 32B LDS reads, one per 16-col tile
            const __bf16* bb = Bp + (kc << 11) + (lane << 4);
            v16bf b0 = *(const v16bf*)(bb);
            v16bf b1 = *(const v16bf*)(bb + 512);
            v16bf b2 = *(const v16bf*)(bb + 1024);
            v16bf b3 = *(const v16bf*)(bb + 1536);
            acc0 = __builtin_amdgcn_wmma_f32_16x16x32_bf16(false, a, false, b0, (short)0, acc0, false, false);
            acc1 = __builtin_amdgcn_wmma_f32_16x16x32_bf16(false, a, false, b1, (short)0, acc1, false, false);
            acc2 = __builtin_amdgcn_wmma_f32_16x16x32_bf16(false, a, false, b2, (short)0, acc2, false, false);
            acc3 = __builtin_amdgcn_wmma_f32_16x16x32_bf16(false, a, false, b3, (short)0, acc3, false, false);
        }
        float* C = feat + (size_t)m * NN * FF;
        const int rbase = rowTile * 16 + (lane >> 4) * 8;
        const int cbase = panel * 64 + (lane & 15);
#pragma unroll
        for (int v = 0; v < 8; ++v) {
            float* cp = C + (size_t)(rbase + v) * FF + cbase;
            cp[0]  = acc0[v];
            cp[16] = acc1[v];
            cp[32] = acc2[v];
            cp[48] = acc3[v];
        }
    }
}

// ---------------------------------------------------------------------------
// K2: el/er[m][n][h] = sum_d feat[m][n][h*32+d] * attn_{l,r}[m][h][d]
// ---------------------------------------------------------------------------
__global__ void node_attn(const float* __restrict__ feat,
                          const float* __restrict__ attn_l,
                          const float* __restrict__ attn_r,
                          float* __restrict__ el, float* __restrict__ er) {
    long gid = (long)blockIdx.x * blockDim.x + threadIdx.x;   // over MM*NN*HH
    if (gid >= (long)MM * NN * HH) return;
    int  hh = (int)(gid & 7);
    long nm = gid >> 3;
    int  n  = (int)(nm % NN);
    int  m  = (int)(nm / NN);
    const float* f  = feat + ((size_t)m * NN + n) * FF + hh * DD;
    const float* al = attn_l + ((size_t)m * HH + hh) * DD;
    const float* ar = attn_r + ((size_t)m * HH + hh) * DD;
    float sl = 0.f, sr = 0.f;
#pragma unroll 8
    for (int d = 0; d < DD; ++d) { float x = f[d]; sl += x * al[d]; sr += x * ar[d]; }
    el[gid] = sl;
    er[gid] = sr;
}

// ---------------------------------------------------------------------------
// K3: segment max of leaky_relu(el[src]+er[dst]) into nmax (encoded uint)
// ---------------------------------------------------------------------------
__global__ void edge_max(const int* __restrict__ src, const int* __restrict__ dst,
                         const float* __restrict__ el, const float* __restrict__ er,
                         unsigned* __restrict__ nmax) {
    long gid = (long)blockIdx.x * blockDim.x + threadIdx.x;   // over MM*EE
    if (gid >= (long)MM * EE) return;
    int e = (int)(gid % EE), m = (int)(gid / EE);
    int s = src[(size_t)m * EE + e], d = dst[(size_t)m * EE + e];
    const float* pl = el + ((size_t)m * NN + s) * HH;
    const float* pr = er + ((size_t)m * NN + d) * HH;
    unsigned*    pm = nmax + ((size_t)m * NN + d) * HH;
#pragma unroll
    for (int hh = 0; hh < HH; ++hh)
        atomicMax(&pm[hh], fenc(lrelu(pl[hh] + pr[hh])));
}

// ---------------------------------------------------------------------------
// K4: segment sum of exp(e - max) into nsum
// ---------------------------------------------------------------------------
__global__ void edge_sum(const int* __restrict__ src, const int* __restrict__ dst,
                         const float* __restrict__ el, const float* __restrict__ er,
                         const unsigned* __restrict__ nmax, float* __restrict__ nsum) {
    long gid = (long)blockIdx.x * blockDim.x + threadIdx.x;
    if (gid >= (long)MM * EE) return;
    int e = (int)(gid % EE), m = (int)(gid / EE);
    int s = src[(size_t)m * EE + e], d = dst[(size_t)m * EE + e];
    const float*    pl = el + ((size_t)m * NN + s) * HH;
    const float*    pr = er + ((size_t)m * NN + d) * HH;
    const unsigned* pm = nmax + ((size_t)m * NN + d) * HH;
    float*          ps = nsum + ((size_t)m * NN + d) * HH;
#pragma unroll
    for (int hh = 0; hh < HH; ++hh) {
        float x = lrelu(pl[hh] + pr[hh]);
        atomicAdd(&ps[hh], __expf(x - fdec(pm[hh])));
    }
}

// ---------------------------------------------------------------------------
// K5: wave-per-edge message aggregation.
// Lane l moves float4 at index l (feat[src][4l..4l+3], head l/8) and at
// index 32+l (head 4+l/8). alpha for head h lives in lane h (h = lane&7);
// broadcast with __shfl. Coalesced 128B gathers; per-lane f32 atomic adds.
// feat (51MB/metapath) + zbuf accumulators stay resident in the 192MB L2.
// ---------------------------------------------------------------------------
__global__ void edge_aggr(const int* __restrict__ src, const int* __restrict__ dst,
                          const float* __restrict__ el, const float* __restrict__ er,
                          const unsigned* __restrict__ nmax, const float* __restrict__ nsum,
                          const float* __restrict__ feat, float* __restrict__ zbuf) {
    const int lane = threadIdx.x & 31;
    long wid = ((long)blockIdx.x * blockDim.x + threadIdx.x) >> 5;  // over MM*EE
    if (wid >= (long)MM * EE) return;
    int e = (int)(wid % EE), m = (int)(wid / EE);
    int s = src[(size_t)m * EE + e], d = dst[(size_t)m * EE + e];

    const int hh = lane & 7;                  // lanes 0..7 own heads 0..7
    size_t nbD = ((size_t)m * NN + d) * HH + hh;
    size_t nbS = ((size_t)m * NN + s) * HH + hh;
    float x  = lrelu(el[nbS] + er[nbD]);
    float mx = fdec(nmax[nbD]);
    float alpha = __expf(x - mx) / fmaxf(nsum[nbD], 1e-38f);
    float a1 = __shfl(alpha, (lane >> 3), 32);        // head lane/8     (0..3)
    float a2 = __shfl(alpha, 4 + (lane >> 3), 32);    // head 4+lane/8   (4..7)

    const float4* fs = (const float4*)(feat + ((size_t)m * NN + s) * FF);
    float*        zp = zbuf + ((size_t)m * NN + d) * FF;
    float4 v1 = fs[lane];        // floats [4*lane, 4*lane+4)
    float4 v2 = fs[32 + lane];   // floats [128+4*lane, ...)
    int o1 = 4 * lane, o2 = 128 + 4 * lane;
    atomicAdd(&zp[o1 + 0], a1 * v1.x);
    atomicAdd(&zp[o1 + 1], a1 * v1.y);
    atomicAdd(&zp[o1 + 2], a1 * v1.z);
    atomicAdd(&zp[o1 + 3], a1 * v1.w);
    atomicAdd(&zp[o2 + 0], a2 * v2.x);
    atomicAdd(&zp[o2 + 1], a2 * v2.y);
    atomicAdd(&zp[o2 + 2], a2 * v2.z);
    atomicAdd(&zp[o2 + 3], a2 * v2.w);
}

// ---------------------------------------------------------------------------
// K6: z_m = elu(rst + bias) in place
// ---------------------------------------------------------------------------
__global__ void bias_elu(float* __restrict__ zbuf, const float* __restrict__ bias) {
    long gid = (long)blockIdx.x * blockDim.x + threadIdx.x;   // over MM*NN*FF
    if (gid >= (long)MM * NN * FF) return;
    int f = (int)(gid % FF);
    int m = (int)(gid / ((long)NN * FF));
    float x = zbuf[gid] + bias[(size_t)m * FF + f];
    zbuf[gid] = x > 0.f ? x : (__expf(x) - 1.f);
}

// ---------------------------------------------------------------------------
// K7: semantic attention logits: wacc[m] += tanh(z_m[n]·W1 + b1)·W2
// Wave per (m,n); z row staged in LDS; lane owns 4 hidden units (float4 W1
// rows are fully coalesced: 512B per k across the wave); shuffle reduction.
// ---------------------------------------------------------------------------
__global__ void semantic(const float* __restrict__ zbuf,
                         const float* __restrict__ w1, const float* __restrict__ b1,
                         const float* __restrict__ w2, float* __restrict__ wacc) {
    __shared__ float zrow[8][FF];
    const int wave = threadIdx.x >> 5, lane = threadIdx.x & 31;
    long wid = (long)blockIdx.x * 8 + wave;   // over MM*NN (150000, exact)
    int n = (int)(wid % NN), m = (int)(wid / NN);

    const float* z = zbuf + ((size_t)m * NN + n) * FF;
    for (int i = lane; i < FF; i += 32) zrow[wave][i] = z[i];  // same wave, no barrier

    float4 q = ((const float4*)b1)[lane];       // hidden units 4*lane .. 4*lane+3
    for (int k = 0; k < FF; ++k) {
        float zk = zrow[wave][k];
        float4 wv = ((const float4*)(w1 + (size_t)k * HIDN))[lane];
        q.x += zk * wv.x; q.y += zk * wv.y; q.z += zk * wv.z; q.w += zk * wv.w;
    }
    float4 w2v = ((const float4*)w2)[lane];
    float part = tanhf(q.x) * w2v.x + tanhf(q.y) * w2v.y +
                 tanhf(q.z) * w2v.z + tanhf(q.w) * w2v.w;
#pragma unroll
    for (int off = 16; off > 0; off >>= 1) part += __shfl_down(part, off, 32);
    if (lane == 0) atomicAdd(&wacc[m], part);
}

// K8: 3-way softmax of the mean logits
__global__ void sem_softmax(const float* __restrict__ wacc, float* __restrict__ aw) {
    if (threadIdx.x == 0 && blockIdx.x == 0) {
        float w0 = wacc[0] / (float)NN, w1 = wacc[1] / (float)NN, w2 = wacc[2] / (float)NN;
        float mx = fmaxf(w0, fmaxf(w1, w2));
        float e0 = __expf(w0 - mx), e1 = __expf(w1 - mx), e2 = __expf(w2 - mx);
        float s = e0 + e1 + e2;
        aw[0] = e0 / s; aw[1] = e1 / s; aw[2] = e2 / s;
    }
}

// K9: z = sum_m aw[m] * z_m   (float4 streaming)
__global__ void combine(const float* __restrict__ zbuf, const float* __restrict__ aw,
                        float* __restrict__ out) {
    long gid = (long)blockIdx.x * blockDim.x + threadIdx.x;   // over NN*FF/4
    if (gid >= (long)NN * (FF / 4)) return;
    const size_t stride = (size_t)NN * (FF / 4);
    float a0 = aw[0], a1 = aw[1], a2 = aw[2];
    const float4* z = (const float4*)zbuf;
    float4 v0 = z[gid], v1 = z[stride + gid], v2 = z[2 * stride + gid];
    float4 r;
    r.x = a0 * v0.x + a1 * v1.x + a2 * v2.x;
    r.y = a0 * v0.y + a1 * v1.y + a2 * v2.y;
    r.z = a0 * v0.z + a1 * v1.z + a2 * v2.z;
    r.w = a0 * v0.w + a1 * v1.w + a2 * v2.w;
    ((float4*)out)[gid] = r;
}

// ---------------------------------------------------------------------------
extern "C" void kernel_launch(void* const* d_in, const int* in_sizes, int n_in,
                              void* d_out, int out_size, void* d_ws, size_t ws_size,
                              hipStream_t stream) {
    const float* h      = (const float*)d_in[0];
    const int*   src    = (const int*)  d_in[1];
    const int*   dst    = (const int*)  d_in[2];
    const float* fcw    = (const float*)d_in[3];
    const float* attn_l = (const float*)d_in[4];
    const float* attn_r = (const float*)d_in[5];
    const float* bias   = (const float*)d_in[6];
    const float* sw1    = (const float*)d_in[7];
    const float* sb1    = (const float*)d_in[8];
    const float* sw2    = (const float*)d_in[9];
    float* out = (float*)d_out;

    // workspace carve-up (floats)
    const size_t ZT = (size_t)MM * NN * FF;   // 38,400,000
    const size_t NH = (size_t)MM * NN * HH;   //  1,200,000
    float*    feat = (float*)d_ws;
    float*    zbuf = feat + ZT;
    float*    el   = zbuf + ZT;
    float*    er   = el + NH;
    unsigned* nmax = (unsigned*)(er + NH);
    float*    nsum = (float*)(nmax + NH);
    float*    wacc = nsum + NH;
    float*    aw   = wacc + 4;
    __bf16*   hb   = (__bf16*)(aw + 4);       // NN*FF bf16 = 25.6MB

    // zero accumulators: nmax(uint 0 == encoded -inf-ish), nsum, wacc, zbuf
    hipMemsetAsync(zbuf, 0, ZT * sizeof(float), stream);
    hipMemsetAsync(nmax, 0, (2 * NH + 8) * sizeof(unsigned), stream);

    // 0b) h -> bf16 once
    conv_h_bf16<<<(int)(((long)NN * FF / 4 + 255) / 256), 256, 0, stream>>>(h, hb);
    // 1) dense projections via WMMA: grid = (row groups, col panels, metapaths)
    gemm_feat_wmma<<<dim3(391, 4, 3), 256, 0, stream>>>(hb, fcw, feat);
    // 2) el/er
    node_attn<<<(int)((NH + 255) / 256), 256, 0, stream>>>(feat, attn_l, attn_r, el, er);
    // 3) edge softmax max
    const long ME = (long)MM * EE;
    edge_max<<<(int)((ME + 255) / 256), 256, 0, stream>>>(src, dst, el, er, nmax);
    // 4) edge softmax denom
    edge_sum<<<(int)((ME + 255) / 256), 256, 0, stream>>>(src, dst, el, er, nmax, nsum);
    // 5) wave-per-edge aggregation (2.4M waves)
    edge_aggr<<<(int)((ME * 32 + 255) / 256), 256, 0, stream>>>(src, dst, el, er,
                                                               nmax, nsum, feat, zbuf);
    // 6) bias + elu
    bias_elu<<<(int)((ZT + 255) / 256), 256, 0, stream>>>(zbuf, bias);
    // 7) semantic logits
    semantic<<<18750, 256, 0, stream>>>(zbuf, sw1, sb1, sw2, wacc);
    // 8) softmax over metapaths
    sem_softmax<<<1, 32, 0, stream>>>(wacc, aw);
    // 9) weighted combine -> output [N, 256]
    combine<<<(int)(((long)NN * (FF / 4) + 255) / 256), 256, 0, stream>>>(zbuf, aw, out);
}
